// TransMaskGenerator_66623532695839
// MI455X (gfx1250) — compile-verified, matrix operand order checked by reading, not run on hardware
//
#include <hip/hip_runtime.h>
#include <hip/hip_bf16.h>

typedef __attribute__((ext_vector_type(16))) _Float16 v16h;
typedef __attribute__((ext_vector_type(8)))  float    v8f;

#define NVOX 262144   // 64*64*64
#define DIM  64

// ---------------------------------------------------------------------------
// Stage 1: voxels = sigmoid(encoded @ W + b)^8, computed with f16 WMMA tiles.
// One wave -> 16(M) x 64(N) of C, K = 256 in 8 steps of 32.
// A layout (16-bit, 16x32): lane l (row = l&15), halves 0..7 -> K = kb + 8*(l>>4) + h,
//                           halves 8..15 -> K = kb + 8*(l>>4) + 16 + (h-8).
// B layout (16-bit, 32x16): mirrored (col = l&15, same K mapping).
// C/D (f32 16x16): lane l holds col (l&15); VGPR r holds row r + 8*(l>>4).
// ---------------------------------------------------------------------------
__global__ __launch_bounds__(256) void voxel_gemm_kernel(
    const float* __restrict__ enc,   // [64,256]
    const float* __restrict__ W,     // [256, NVOX]
    const float* __restrict__ bias,  // [NVOX]
    float* __restrict__ voxels)      // [64, NVOX]
{
    const int wave  = blockIdx.x * (blockDim.x >> 5) + (threadIdx.x >> 5);
    const int lane  = threadIdx.x & 31;
    const int mTile = wave & 3;          // 4 m-tiles of 16 rows (B=64)
    const int nTile = wave >> 2;         // 4096 n-tiles of 64 cols
    const int m0 = mTile * 16;
    const int n0 = nTile * 64;
    const int hs = lane >> 4;            // lane-half select
    const int lr = lane & 15;

    v8f acc0 = {}, acc1 = {}, acc2 = {}, acc3 = {};
    const float* aBase = enc + (m0 + lr) * 256 + hs * 8;

    for (int kb = 0; kb < 256; kb += 32) {
        // ---- A tile (16x32 f16), two contiguous 8-float runs per lane ----
        v16h a;
#pragma unroll
        for (int h = 0; h < 8; ++h)  a[h]     = (_Float16)aBase[kb + h];
#pragma unroll
        for (int h = 0; h < 8; ++h)  a[8 + h] = (_Float16)aBase[kb + 16 + h];

        // ---- 4 B tiles (32x16 f16 each), W streamed non-temporally ----
#pragma unroll
        for (int t = 0; t < 4; ++t) {
            const int col = n0 + t * 16 + lr;
            const float* wB = W + (size_t)(kb + hs * 8) * NVOX + col;
            v16h bm;
#pragma unroll
            for (int h = 0; h < 8; ++h)
                bm[h]     = (_Float16)__builtin_nontemporal_load(wB + (size_t)h * NVOX);
#pragma unroll
            for (int h = 0; h < 8; ++h)
                bm[8 + h] = (_Float16)__builtin_nontemporal_load(wB + (size_t)(16 + h) * NVOX);

            if (t == 0) acc0 = __builtin_amdgcn_wmma_f32_16x16x32_f16(false, a, false, bm, (short)0, acc0, false, false);
            if (t == 1) acc1 = __builtin_amdgcn_wmma_f32_16x16x32_f16(false, a, false, bm, (short)0, acc1, false, false);
            if (t == 2) acc2 = __builtin_amdgcn_wmma_f32_16x16x32_f16(false, a, false, bm, (short)0, acc2, false, false);
            if (t == 3) acc3 = __builtin_amdgcn_wmma_f32_16x16x32_f16(false, a, false, bm, (short)0, acc3, false, false);
        }
    }

    // ---- epilogue: bias + sigmoid + ^8 (ALPHA_ADJ == 8), store voxels ----
#pragma unroll
    for (int t = 0; t < 4; ++t) {
        const v8f acc = (t == 0) ? acc0 : (t == 1) ? acc1 : (t == 2) ? acc2 : acc3;
        const int n = n0 + t * 16 + lr;
        const float bv = bias[n];
#pragma unroll
        for (int r = 0; r < 8; ++r) {
            const int m = m0 + r + hs * 8;
            const float x  = acc[r] + bv;
            const float s  = 1.0f / (1.0f + __expf(-x));
            const float s2 = s * s;
            const float s4 = s2 * s2;
            voxels[(size_t)m * NVOX + n] = s4 * s4;
        }
    }
}

// ---------------------------------------------------------------------------
// Stage 2: per-pixel rotated trilinear ray march + back-to-front compositing.
// views[b,i,j,k] = trilinear(vox[b], R @ (dst - c) + c); img scanned k=63..0.
// Voxel array (64 MB) lives in L2 (192 MB) so the 8 gathers/step stay on-chip.
// ---------------------------------------------------------------------------
__global__ __launch_bounds__(256) void render_kernel(
    const float* __restrict__ voxels,  // [64, NVOX]
    const float* __restrict__ angles,  // [64, 3]
    float* __restrict__ out)           // [64, 1, 64, 64]
{
    const int tid = blockIdx.x * blockDim.x + threadIdx.x;
    const int b  = tid >> 12;
    const int ij = tid & 4095;
    const int i  = ij >> 6;
    const int j  = ij & 63;

    const float ax = angles[b * 3 + 0];
    const float ay = angles[b * 3 + 1];
    const float az = angles[b * 3 + 2];
    const float cx = __cosf(ax), sx = __sinf(ax);
    const float cy = __cosf(ay), sy = __sinf(ay);
    const float cz = __cosf(az), sz = __sinf(az);
    // R = Rz @ Ry @ Rx
    const float r00 = cz * cy, r01 = cz * sy * sx - sz * cx, r02 = cz * sy * cx + sz * sx;
    const float r10 = sz * cy, r11 = sz * sy * sx + cz * cx, r12 = sz * sy * cx - cz * sx;
    const float r20 = -sy,     r21 = cy * sx,                r22 = cy * cx;

    const float c  = (DIM - 1) * 0.5f;   // 31.5
    const float di = (float)i - c;
    const float dj = (float)j - c;
    const float bx = r00 * di + r01 * dj + c;
    const float by = r10 * di + r11 * dj + c;
    const float bz = r20 * di + r21 * dj + c;

    const float* __restrict__ vol = voxels + (size_t)b * NVOX;

    float img = 0.0f;
    for (int k = DIM - 1; k >= 0; --k) {
        const float dk = (float)k - c;
        const float x = bx + r02 * dk;
        const float y = by + r12 * dk;
        const float z = bz + r22 * dk;
        const float x0 = floorf(x), y0 = floorf(y), z0 = floorf(z);
        const float fx = x - x0, fy = y - y0, fz = z - z0;
        const int ix = (int)x0, iy = (int)y0, iz = (int)z0;

        auto g = [&](int X, int Y, int Z) -> float {
            if ((unsigned)X < (unsigned)DIM && (unsigned)Y < (unsigned)DIM &&
                (unsigned)Z < (unsigned)DIM)
                return vol[(X << 12) + (Y << 6) + Z];
            return 0.0f;
        };
        const float c000 = g(ix,     iy,     iz    );
        const float c001 = g(ix,     iy,     iz + 1);
        const float c010 = g(ix,     iy + 1, iz    );
        const float c011 = g(ix,     iy + 1, iz + 1);
        const float c100 = g(ix + 1, iy,     iz    );
        const float c101 = g(ix + 1, iy,     iz + 1);
        const float c110 = g(ix + 1, iy + 1, iz    );
        const float c111 = g(ix + 1, iy + 1, iz + 1);

        const float c00 = c000 * (1.0f - fz) + c001 * fz;
        const float c01 = c010 * (1.0f - fz) + c011 * fz;
        const float c10 = c100 * (1.0f - fz) + c101 * fz;
        const float c11 = c110 * (1.0f - fz) + c111 * fz;
        const float c0  = c00 * (1.0f - fy) + c01 * fy;
        const float c1  = c10 * (1.0f - fy) + c11 * fy;
        const float a   = c0 * (1.0f - fx) + c1 * fx;

        img = img * (1.0f - a) + a;       // back-to-front composite
    }
    out[tid] = 2.0f * img - 1.0f;
}

extern "C" void kernel_launch(void* const* d_in, const int* in_sizes, int n_in,
                              void* d_out, int out_size, void* d_ws, size_t ws_size,
                              hipStream_t stream) {
    const float* enc    = (const float*)d_in[0];  // [64,256]
    const float* W      = (const float*)d_in[1];  // [256,262144]
    const float* bias   = (const float*)d_in[2];  // [262144]
    const float* angles = (const float*)d_in[3];  // [64,3]
    float* out    = (float*)d_out;                // [64*64*64]
    float* voxels = (float*)d_ws;                 // 64*262144*4 = 64 MiB scratch

    // Stage 1: 16384 waves (4 m-tiles x 4096 n-tiles), 8 waves/block.
    voxel_gemm_kernel<<<2048, 256, 0, stream>>>(enc, W, bias, voxels);
    // Stage 2: one thread per (b, i, j) = 262144 threads.
    render_kernel<<<1024, 256, 0, stream>>>(voxels, angles, out);
}
